// TransLayer_35098472743203
// MI455X (gfx1250) — matrix operand order repaired and checked by reading, not compiled, not run on hardware
//
#include <hip/hip_runtime.h>
#include <math.h>

typedef __bf16 bf16_t;
typedef __attribute__((ext_vector_type(16))) __bf16 v16bf;
typedef __attribute__((ext_vector_type(8)))  __bf16 v8bf;
typedef __attribute__((ext_vector_type(8)))  float  v8f;
typedef __attribute__((ext_vector_type(4)))  int    v4i;

// ---------- CDNA5 async global->LDS (ASYNCcnt path), with safe fallback ----
#if defined(__has_builtin)
#if __has_builtin(__builtin_amdgcn_global_load_async_to_lds_b128) && \
    __has_builtin(__builtin_amdgcn_s_wait_asynccnt)
#define HAVE_ASYNC 1
#endif
#endif
#ifndef HAVE_ASYNC
#define HAVE_ASYNC 0
#endif

#if HAVE_ASYNC
typedef __attribute__((address_space(1))) v4i g_v4i;
typedef __attribute__((address_space(3))) v4i l_v4i;
#define ASYNC_COPY16(dst, src)                                              \
  __builtin_amdgcn_global_load_async_to_lds_b128(                           \
      (g_v4i*)(src), (l_v4i*)(dst), 0, 0)
#define ASYNC_WAIT(n) __builtin_amdgcn_s_wait_asynccnt(n)
#else
#define ASYNC_COPY16(dst, src)                                              \
  do {                                                                      \
    *(v8bf*)(dst) = *(const v8bf*)(src);                                    \
    *(v8bf*)((bf16_t*)(dst) + 8) = *(const v8bf*)((const bf16_t*)(src) + 8);\
  } while (0)
#define ASYNC_WAIT(n) do {} while (0)
#endif

__device__ __forceinline__ int imin(int a, int b) { return a < b ? a : b; }

// ---------------- f32 -> bf16 with column padding ----------------
__global__ __launch_bounds__(256) void cvt_pad_kernel(
    const float* __restrict__ in, bf16_t* __restrict__ out,
    int R, int C, int Cp) {
  long long idx = (long long)blockIdx.x * 256 + threadIdx.x;
  long long total = (long long)R * Cp;
  if (idx >= total) return;
  int r = (int)(idx / Cp);
  int c = (int)(idx % Cp);
  float v = (c < C) ? in[(long long)r * C + c] : 0.0f;
  out[idx] = (bf16_t)v;
}

// ---------------- row softmax: f32 scores -> bf16 probs ----------------
__global__ __launch_bounds__(256) void softmax_kernel(
    const float* __restrict__ S, bf16_t* __restrict__ P, int n, float scale) {
  __shared__ float red[256];
  const int row = blockIdx.x;
  const long long base = (long long)row * n;
  const int tid = threadIdx.x;
  float lmax = -3.4e38f;
  for (int c = tid; c < n; c += 256) lmax = fmaxf(lmax, S[base + c] * scale);
  red[tid] = lmax; __syncthreads();
  for (int s = 128; s > 0; s >>= 1) {
    if (tid < s) red[tid] = fmaxf(red[tid], red[tid + s]);
    __syncthreads();
  }
  const float m = red[0]; __syncthreads();
  float lsum = 0.f;
  for (int c = tid; c < n; c += 256) lsum += expf(S[base + c] * scale - m);
  red[tid] = lsum; __syncthreads();
  for (int s = 128; s > 0; s >>= 1) {
    if (tid < s) red[tid] += red[tid + s];
    __syncthreads();
  }
  const float inv = 1.0f / red[0];
  for (int c = tid; c < n; c += 256)
    P[base + c] = (bf16_t)(expf(S[base + c] * scale - m) * inv);
}

// ---------------- bf16 WMMA GEMM, double-buffered & pipelined ------------
// C[m,n] = sum_k A[m,k] * Bmat[k,n]   (+ epilogue)
// BMODE 0 (NT): Bg [Brows, ldb] row-major, Bmat[k][n] = Bg[n][k]
// BMODE 1 (NN): Bg [K, ldb]     row-major, Bmat[k][n] = Bg[k][n]
// EPI 0: bf16 +bias | EPI 1: bf16 gelu(x+bias) | EPI 2: f32 | EPI 3: f32 +bias +resid
// Block: 256 threads = 8 waves (2 x 4); block tile 128x128; wave tile 64x32.
template<int BMODE, int EPI>
__global__ __launch_bounds__(256) void gemm_bf16_wmma(
    const bf16_t* __restrict__ A,  long long sA, int lda,
    const bf16_t* __restrict__ Bg, long long sB, int ldb, int Brows, int Bcols,
    void* __restrict__ Outv, long long sO, int ldo,
    const float* __restrict__ bias,
    const float* __restrict__ resid,
    int Evalid, int Ksteps) {
  constexpr int BM = 128, BN = 128, BK = 32, LDP = BK + 8;
  __shared__ bf16_t As[2][BM][LDP];
  __shared__ bf16_t Bs[2][BN][LDP];

  const int tid = threadIdx.x;
  const int b = blockIdx.z;
  A  += (long long)b * sA;
  Bg += (long long)b * sB;
  const long long obase = (long long)b * sO;
  const int m0 = blockIdx.x * BM;
  const int n0 = blockIdx.y * BN;

  const int wave = tid >> 5;
  const int lane = tid & 31;
  const int wm = wave >> 2;   // 0..1 : 64-row strip
  const int wn = wave & 3;    // 0..3 : 32-col strip
  const int lh = lane & 15;
  const int lg = lane >> 4;   // 0/1

  // per-wave async copy geometry: 8 rows per b128 instr (4 lanes/row)
  const int cprow = lane >> 2;         // 0..7
  const int cpkc  = (lane & 3) * 8;    // 0,8,16,24 halves

  auto issueA = [&](int ks, int buf) {
    const int kg0 = ks * BK;
#pragma unroll
    for (int q = 0; q < 2; ++q) {
      const int r = wave * 16 + q * 8 + cprow;
      ASYNC_COPY16(&As[buf][r][cpkc],
                   A + (long long)(m0 + r) * lda + kg0 + cpkc);
    }
  };
  auto issueBnt = [&](int ks, int buf) {
    const int kg0 = ks * BK;
#pragma unroll
    for (int q = 0; q < 2; ++q) {
      const int r = wave * 16 + q * 8 + cprow;
      const int e = imin(n0 + r, Brows - 1);   // clamped: garbage lands in cols >= Evalid
      ASYNC_COPY16(&Bs[buf][r][cpkc],
                   Bg + (long long)e * ldb + kg0 + cpkc);
    }
  };
  // NN B-tile: register-staged (needs transpose scatter into Bs[n][k])
  const int nnK  = tid & 31;
  const int nnNb = (tid >> 5) * 16;
  v8bf nnB0 = {}, nnB1 = {};
  auto loadBnnRegs = [&](int ks) {
    const int kr = imin(ks * BK + nnK, Brows - 1);
    const int cn = imin(n0 + nnNb, Bcols - 16);  // clamped: only affects cols >= Evalid
    const bf16_t* src = Bg + (long long)kr * ldb + cn;
    nnB0 = *(const v8bf*)(src);
    nnB1 = *(const v8bf*)(src + 8);
  };
  auto storeBnn = [&](int buf) {
#pragma unroll
    for (int j = 0; j < 8; ++j) Bs[buf][nnNb + j][nnK] = nnB0[j];
#pragma unroll
    for (int j = 0; j < 8; ++j) Bs[buf][nnNb + 8 + j][nnK] = nnB1[j];
  };

  v8f c[4][2] = {};

  // ---- prologue: tile 0 into buffer 0
  issueA(0, 0);
  if (BMODE == 0) {
    issueBnt(0, 0);
  } else {
    loadBnnRegs(0);
    storeBnn(0);
  }

  for (int ks = 0; ks < Ksteps; ++ks) {
    const int cur = ks & 1, nxt = cur ^ 1;
    const bool more = (ks + 1 < Ksteps);
    if (more) {               // buf[nxt] last read before the previous barrier -> safe
      issueA(ks + 1, nxt);
      if (BMODE == 0) issueBnt(ks + 1, nxt);
      else            loadBnnRegs(ks + 1);
    }
    // retire this tile's async copies, then make all waves' data visible
    if (more) ASYNC_WAIT(BMODE == 0 ? 4 : 2);
    else      ASYNC_WAIT(0);
    __syncthreads();

    // ---- fragments (ISA 7.12.2 bf16 layouts) + 8 WMMA
    v16bf afrag[4], bfrag[2];
#pragma unroll
    for (int i = 0; i < 4; ++i) {
      const int row = wm * 64 + i * 16 + lh;
      v8bf lo = *(const v8bf*)&As[cur][row][lg * 8];
      v8bf hi = *(const v8bf*)&As[cur][row][16 + lg * 8];
      afrag[i] = __builtin_shufflevector(lo, hi, 0,1,2,3,4,5,6,7,8,9,10,11,12,13,14,15);
    }
#pragma unroll
    for (int j = 0; j < 2; ++j) {
      const int col = wn * 32 + j * 16 + lh;
      v8bf lo = *(const v8bf*)&Bs[cur][col][lg * 8];
      v8bf hi = *(const v8bf*)&Bs[cur][col][16 + lg * 8];
      bfrag[j] = __builtin_shufflevector(lo, hi, 0,1,2,3,4,5,6,7,8,9,10,11,12,13,14,15);
    }
#pragma unroll
    for (int i = 0; i < 4; ++i)
#pragma unroll
      for (int j = 0; j < 2; ++j)
        c[i][j] = __builtin_amdgcn_wmma_f32_16x16x32_bf16(
            false, afrag[i], false, bfrag[j], (short)0, c[i][j], false, false);

    if (BMODE == 1 && more) storeBnn(nxt);
    __syncthreads();          // reads of buf[cur] done; NN stores visible
  }

  // ---- epilogue (C/D layout: VGPR v -> M = v + 8*lg ; N = lane%16)
#pragma unroll
  for (int i = 0; i < 4; ++i) {
#pragma unroll
    for (int j = 0; j < 2; ++j) {
      const int col = n0 + wn * 32 + j * 16 + lh;
      if (col >= ldo) continue;
      float bcol = 0.f;
      if (bias != nullptr && col < Evalid) bcol = bias[col];
#pragma unroll
      for (int v = 0; v < 8; ++v) {
        const int row = m0 + wm * 64 + i * 16 + lg * 8 + v;
        float val;
        if (col >= Evalid) {
          val = 0.f;
        } else {
          val = c[i][j][v] + bcol;
          if (EPI == 1) val = 0.5f * val * (1.0f + erff(val * 0.70710678118654752f));
        }
        const long long oidx = obase + (long long)row * ldo + col;
        if (EPI == 2 || EPI == 3) {
          float o = val;
          if (EPI == 3 && resid != nullptr && col < Evalid)
            o += resid[(long long)row * ldo + col];
          ((float*)Outv)[oidx] = o;
        } else {
          ((bf16_t*)Outv)[oidx] = (bf16_t)val;
        }
      }
    }
  }
}

extern "C" void kernel_launch(void* const* d_in, const int* in_sizes, int n_in,
                              void* d_out, int out_size, void* d_ws, size_t ws_size,
                              hipStream_t stream) {
  (void)in_sizes; (void)n_in; (void)out_size; (void)ws_size;
  constexpr int D  = 1028, Kp = 1056;           // 1056 = 33*32
  constexpr int H  = 4 * D /*4112*/, Hp = 4128; // 4128 = 129*32
  constexpr int Bb = 8, N = 2048, M = Bb * N;   // 16384 rows

  const float* x  = (const float*)d_in[0];
  const float* Wq = (const float*)d_in[1];
  const float* bq = (const float*)d_in[2];
  const float* Wk = (const float*)d_in[3];
  const float* bk = (const float*)d_in[4];
  const float* Wv = (const float*)d_in[5];
  const float* bv = (const float*)d_in[6];
  const float* We = (const float*)d_in[7];
  const float* be = (const float*)d_in[8];
  const float* Wo = (const float*)d_in[9];
  const float* bo = (const float*)d_in[10];
  float* out = (float*)d_out;

  // ---- workspace layout (aliased regions)
  char* ws = (char*)d_ws;
  size_t off = 0;
  auto alloc = [&](size_t bytes) {
    off = (off + 255) & ~(size_t)255;
    size_t r = off; off += bytes; return r;
  };
  const size_t o_xb  = alloc((size_t)M * Kp * 2);     // x bf16; later reused as attn
  const size_t o_wqb = alloc((size_t)D * Kp * 2);
  const size_t o_wkb = alloc((size_t)D * Kp * 2);
  const size_t o_wvb = alloc((size_t)D * Kp * 2);
  const size_t o_web = alloc((size_t)H * Kp * 2);
  const size_t o_wob = alloc((size_t)D * Hp * 2);
  const size_t o_qb  = alloc((size_t)M * Kp * 2);     // q; later start of probs
  const size_t o_kb  = alloc((size_t)M * Kp * 2);     // k; contiguous after q
  const size_t o_vb  = alloc((size_t)M * Kp * 2);
  size_t sscore = (size_t)Bb * N * N * 4;             // 134.2 MB f32 scores
  size_t sh     = (size_t)M * Hp * 2;                 // 135.3 MB bf16 h
  const size_t o_s32 = alloc(sscore > sh ? sscore : sh);

  bf16_t* xb  = (bf16_t*)(ws + o_xb);
  bf16_t* wqb = (bf16_t*)(ws + o_wqb);
  bf16_t* wkb = (bf16_t*)(ws + o_wkb);
  bf16_t* wvb = (bf16_t*)(ws + o_wvb);
  bf16_t* web = (bf16_t*)(ws + o_web);
  bf16_t* wob = (bf16_t*)(ws + o_wob);
  bf16_t* qb  = (bf16_t*)(ws + o_qb);
  bf16_t* kb  = (bf16_t*)(ws + o_kb);
  bf16_t* vb  = (bf16_t*)(ws + o_vb);
  float*  s32 = (float*)(ws + o_s32);
  bf16_t* pb  = (bf16_t*)(ws + o_qb);   // probs alias q||k (67 MB <= 69.2 MB)
  bf16_t* ab  = (bf16_t*)(ws + o_xb);   // attn alias xb
  bf16_t* hb  = (bf16_t*)(ws + o_s32);  // h alias scores

  auto cvt = [&](const float* src, bf16_t* dst, int R, int C, int Cp) {
    long long total = (long long)R * Cp;
    int grid = (int)((total + 255) / 256);
    cvt_pad_kernel<<<grid, 256, 0, stream>>>(src, dst, R, C, Cp);
  };

  // ---- stage bf16 operands (zero-padded K)
  cvt(x,  xb,  M, D, Kp);
  cvt(Wq, wqb, D, D, Kp);
  cvt(Wk, wkb, D, D, Kp);
  cvt(Wv, wvb, D, D, Kp);
  cvt(We, web, H, D, Kp);
  cvt(Wo, wob, D, H, Hp);

  // ---- QKV projections: [M,Kp] @ W^T + b  -> bf16 [M,Kp] (pad cols zeroed)
  dim3 gProj(M / 128, (Kp + 127) / 128, 1);
  gemm_bf16_wmma<0, 0><<<gProj, 256, 0, stream>>>(
      xb, 0, Kp, wqb, 0, Kp, D, 0, qb, 0, Kp, bq, nullptr, D, Kp / 32);
  gemm_bf16_wmma<0, 0><<<gProj, 256, 0, stream>>>(
      xb, 0, Kp, wkb, 0, Kp, D, 0, kb, 0, Kp, bk, nullptr, D, Kp / 32);
  gemm_bf16_wmma<0, 0><<<gProj, 256, 0, stream>>>(
      xb, 0, Kp, wvb, 0, Kp, D, 0, vb, 0, Kp, bv, nullptr, D, Kp / 32);

  // ---- scores = q @ k^T  (per batch), f32
  dim3 gScore(N / 128, N / 128, Bb);
  gemm_bf16_wmma<0, 2><<<gScore, 256, 0, stream>>>(
      qb, (long long)N * Kp, Kp, kb, (long long)N * Kp, Kp, N, 0,
      s32, (long long)N * N, N, nullptr, nullptr, N, Kp / 32);

  // ---- softmax(scores / sqrt(1028)) -> bf16 probs
  softmax_kernel<<<Bb * N, 256, 0, stream>>>(s32, pb, N, 1.0f / sqrtf((float)D));

  // ---- attn = probs @ v  (NN), bf16 [M,Kp]
  dim3 gAttn(N / 128, (Kp + 127) / 128, Bb);
  gemm_bf16_wmma<1, 0><<<gAttn, 256, 0, stream>>>(
      pb, (long long)N * N, N, vb, (long long)N * Kp, Kp, N, Kp,
      ab, (long long)N * Kp, Kp, nullptr, nullptr, Kp, N / 32);

  // ---- h = gelu(attn @ We^T + be), bf16 [M,Hp]
  dim3 gFfn1(M / 128, (Hp + 127) / 128, 1);
  gemm_bf16_wmma<0, 1><<<gFfn1, 256, 0, stream>>>(
      ab, 0, Kp, web, 0, Kp, H, 0, hb, 0, Hp, be, nullptr, H, Kp / 32);

  // ---- out = h @ Wo^T + bo + x, f32 [M,D]
  dim3 gFfn2(M / 128, (D + 127) / 128, 1);
  gemm_bf16_wmma<0, 3><<<gFfn2, 256, 0, stream>>>(
      hb, 0, Hp, wob, 0, Hp, D, 0, out, 0, D, bo, x, D, Hp / 32);
}
